// DiffAttn_14121852469507
// MI455X (gfx1250) — matrix operand order repaired and checked by reading, not compiled, not run on hardware
//
#include <hip/hip_runtime.h>

// ---------------------------------------------------------------------------
// Types for CDNA5 WMMA (wave32): V_WMMA_F32_16X16X32_BF16
// ---------------------------------------------------------------------------
typedef __attribute__((ext_vector_type(16))) __bf16        v16bf;
typedef __attribute__((ext_vector_type(8)))  float         v8f;
typedef __attribute__((ext_vector_type(4)))  unsigned int  v4u;
typedef __attribute__((ext_vector_type(4)))  int           v4i;

union FragU {            // one 16x32 bf16 A-frag or 32x16 bf16 B-frag per lane
    v4u   u[2];          // 2 x 16B loads
    v16bf bf;
};
union U8x16 {            // 8 bf16 elements as one 16B vector
    v4u            u;
    unsigned short s[8];
};

// ---- CDNA5 async global->LDS copy (ASYNCcnt), guarded ---------------------
// Probe round 2 confirmed the builtin exists with params:
//   (int4 __device__ *src, int4 __shared__ *dst, imm int offset, imm int cpol)
#if defined(__has_builtin)
#if __has_builtin(__builtin_amdgcn_global_load_async_to_lds_b128)
#define HAVE_ASYNC_LDS 1
#endif
#endif

#define AS1V(p) ((__attribute__((address_space(1))) v4i*)(p))
#define AS3V(p) ((__attribute__((address_space(3))) v4i*)(p))

__device__ __forceinline__ void wait_async_zero() {
#if defined(HAVE_ASYNC_LDS)
#if __has_builtin(__builtin_amdgcn_s_wait_asynccnt)
    __builtin_amdgcn_s_wait_asynccnt(0);
#else
    asm volatile("s_wait_asynccnt 0x0" ::: "memory");
#endif
#endif
}

__device__ __forceinline__ unsigned short f32_to_bf16(float f) {
    unsigned int u = __float_as_uint(f);
    unsigned int r = u + 0x7FFFu + ((u >> 16) & 1u);   // round-to-nearest-even
    return (unsigned short)(r >> 16);
}

// ---------------------------------------------------------------------------
// Stage 0a: X fp32 -> bf16
// ---------------------------------------------------------------------------
__global__ void cvt_x_kernel(const float* __restrict__ X,
                             unsigned short* __restrict__ Xb, int n) {
    int i = blockIdx.x * blockDim.x + threadIdx.x;
    for (; i < n; i += gridDim.x * blockDim.x) Xb[i] = f32_to_bf16(X[i]);
}

// ---------------------------------------------------------------------------
// Stage 0b: pack Wq(1024x1024) | Wk(1024x1024) | Wv(512x1024) -> bf16 [2560x1024]
//           and bq|bk|bv -> fp32 bias[2560]
// ---------------------------------------------------------------------------
__global__ void pack_w_kernel(const float* __restrict__ Wq, const float* __restrict__ Wk,
                              const float* __restrict__ Wv, const float* __restrict__ bq,
                              const float* __restrict__ bk, const float* __restrict__ bv,
                              unsigned short* __restrict__ Wb, float* __restrict__ biasP) {
    const int n = 2560 * 1024;
    int i = blockIdx.x * blockDim.x + threadIdx.x;
    const int j = i;
    for (; i < n; i += gridDim.x * blockDim.x) {
        int row = i >> 10, col = i & 1023;
        float v;
        if (row < 1024)      v = Wq[row * 1024 + col];
        else if (row < 2048) v = Wk[(row - 1024) * 1024 + col];
        else                 v = Wv[(row - 2048) * 1024 + col];
        Wb[i] = f32_to_bf16(v);
    }
    if (j < 2560)
        biasP[j] = (j < 1024) ? bq[j] : (j < 2048 ? bk[j - 1024] : bv[j - 2048]);
}

// ---------------------------------------------------------------------------
// Stage 1: QKV projection GEMM.  C[16384 x 2560] = Xb @ Wb^T + bias
// BM=128, BN=128, BK=64, double-buffered LDS (2 x 32 KB), async copies.
// 8 waves, each owns a 64x32 sub-tile (4x2 WMMA accumulators).
// ---------------------------------------------------------------------------
__device__ __forceinline__
void gemm_load_tile(const unsigned short* __restrict__ Xb,
                    const unsigned short* __restrict__ Wb,
                    long m0, int n0, int kb, int tid,
                    unsigned short* Xs, unsigned short* Ws) {
#pragma unroll
    for (int i = 0; i < 4; ++i) {              // 1024 chunks of 8 bf16, 4/thread
        int c = tid + 256 * i;
        int row = c >> 3, col = (c & 7) * 8;
#if defined(HAVE_ASYNC_LDS)
        __builtin_amdgcn_global_load_async_to_lds_b128(
            AS1V(Xb + (m0 + row) * 1024 + kb * 64 + col),
            AS3V(Xs + row * 64 + col), 0, 0);
        __builtin_amdgcn_global_load_async_to_lds_b128(
            AS1V(Wb + (long)(n0 + row) * 1024 + kb * 64 + col),
            AS3V(Ws + row * 64 + col), 0, 0);
#else
        *(v4u*)(Xs + row * 64 + col) =
            *(const v4u*)(Xb + (m0 + row) * 1024 + kb * 64 + col);
        *(v4u*)(Ws + row * 64 + col) =
            *(const v4u*)(Wb + (long)(n0 + row) * 1024 + kb * 64 + col);
#endif
    }
}

#define GEMM_SMEM_BYTES (4 * 128 * 64 * 2)     // Xs[2] + Ws[2] = 64 KB

__global__ __launch_bounds__(256)
void qkv_gemm_kernel(const unsigned short* __restrict__ Xb,
                     const unsigned short* __restrict__ Wb,
                     const float* __restrict__ bias,
                     unsigned short* __restrict__ Qb,
                     unsigned short* __restrict__ Kb,
                     unsigned short* __restrict__ Vb) {
    extern __shared__ __align__(16) unsigned short gsmem[];
    // buffer p: Xs = gsmem + p*8192, Ws = gsmem + 16384 + p*8192

    const int tid  = threadIdx.x;
    const int lane = tid & 31;
    const int wave = tid >> 5;
    const int half = lane >> 4;       // 0: lanes 0-15, 1: lanes 16-31
    const int l16  = lane & 15;
    const int wm   = wave >> 2;       // 0..1  -> 64 rows each
    const int wn   = wave & 3;        // 0..3  -> 32 cols each
    const long m0  = (long)blockIdx.x * 128;
    const int  n0  = blockIdx.y * 128;

    v8f acc[4][2];
#pragma unroll
    for (int tm = 0; tm < 4; ++tm)
#pragma unroll
        for (int tn = 0; tn < 2; ++tn)
#pragma unroll
            for (int r = 0; r < 8; ++r) acc[tm][tn][r] = 0.0f;

    gemm_load_tile(Xb, Wb, m0, n0, 0, tid, gsmem, gsmem + 2 * 128 * 64);
    wait_async_zero();
    __syncthreads();

    for (int kb = 0; kb < 16; ++kb) {          // K = 1024 in chunks of 64
        const int p = kb & 1;
        unsigned short* Xs = gsmem + p * (128 * 64);
        unsigned short* Ws = gsmem + 2 * 128 * 64 + p * (128 * 64);
        if (kb + 1 < 16)                       // prefetch next tile into 1-p
            gemm_load_tile(Xb, Wb, m0, n0, kb + 1, tid,
                           gsmem + (1 - p) * (128 * 64),
                           gsmem + 2 * 128 * 64 + (1 - p) * (128 * 64));

#pragma unroll
        for (int ks = 0; ks < 2; ++ks) {       // two K=32 WMMA steps
            FragU A[4], B[2];
#pragma unroll
            for (int tm = 0; tm < 4; ++tm) {   // A: lane=row, k split per ISA layout
                const unsigned short* ptr =
                    Xs + (wm * 64 + tm * 16 + l16) * 64 + ks * 32 + half * 8;
                A[tm].u[0] = *(const v4u*)(ptr);
                A[tm].u[1] = *(const v4u*)(ptr + 16);
            }
#pragma unroll
            for (int tn = 0; tn < 2; ++tn) {   // B: lane=col, k split per ISA layout
                const unsigned short* ptr =
                    Ws + (wn * 32 + tn * 16 + l16) * 64 + ks * 32 + half * 8;
                B[tn].u[0] = *(const v4u*)(ptr);
                B[tn].u[1] = *(const v4u*)(ptr + 16);
            }
#pragma unroll
            for (int tm = 0; tm < 4; ++tm)
#pragma unroll
                for (int tn = 0; tn < 2; ++tn)
                    acc[tm][tn] = __builtin_amdgcn_wmma_f32_16x16x32_bf16(
                        false, A[tm].bf, false, B[tn].bf,
                        (short)0, acc[tm][tn], false, false);
        }
        wait_async_zero();
        __syncthreads();
    }

    // epilogue: +bias, bf16, scatter to Q/K/V regions
#pragma unroll
    for (int tm = 0; tm < 4; ++tm)
#pragma unroll
        for (int tn = 0; tn < 2; ++tn) {
            int n = n0 + wn * 32 + tn * 16 + l16;
            float bvx = bias[n];
#pragma unroll
            for (int r = 0; r < 8; ++r) {
                long m = m0 + wm * 64 + tm * 16 + r + 8 * half;
                unsigned short v = f32_to_bf16(acc[tm][tn][r] + bvx);
                if (n < 1024)      Qb[m * 1024 + n] = v;
                else if (n < 2048) Kb[m * 1024 + (n - 1024)] = v;
                else               Vb[m * 512 + (n - 2048)] = v;
            }
        }
}

// ---------------------------------------------------------------------------
// Stage 2: flash-attention pass (one head slice of D=512).
//   head==0:  out = softmax(Q1 K1^T / sqrt(D)) @ V
//   head==1:  out = out - lambda * softmax(Q2 K2^T / sqrt(D)) @ V
// Block: 128 queries, 8 waves x 16 rows. 64 key tiles of 64, double-buffered:
//   K tiles 2 x 64 KB (async copy) + V^T tiles 2 x 64 KB + P 16 KB = 272 KB LDS.
// ---------------------------------------------------------------------------
__device__ __forceinline__
void attn_load_tile(const unsigned short* __restrict__ Kb,
                    const unsigned short* __restrict__ Vb,
                    long krow0, int head, int tid,
                    unsigned short* Ks, unsigned short* Vt) {
#pragma unroll
    for (int i = 0; i < 16; ++i) {             // 4096 chunks of 8 bf16
        int c   = tid + 256 * i;
        int key = c >> 6;
        int dc  = (c & 63) * 8;
        const long krow = krow0 + key;
#if defined(HAVE_ASYNC_LDS)
        __builtin_amdgcn_global_load_async_to_lds_b128(
            AS1V(Kb + krow * 1024 + (long)head * 512 + dc),
            AS3V(Ks + key * 512 + dc), 0, 0);
#else
        *(v4u*)(Ks + key * 512 + dc) =
            *(const v4u*)(Kb + krow * 1024 + (long)head * 512 + dc);
#endif
        U8x16 vv;                               // V: transposed scatter to LDS
        vv.u = *(const v4u*)(Vb + krow * 512 + dc);
#pragma unroll
        for (int e = 0; e < 8; ++e) Vt[(dc + e) * 64 + key] = vv.s[e];
    }
}

#define ATTN_SMEM_BYTES ((2 * 64 * 512 + 2 * 512 * 64 + 8 * 16 * 64) * 2)  // 272 KB

__global__ __launch_bounds__(256)
void attn_pass_kernel(const unsigned short* __restrict__ Qb,
                      const unsigned short* __restrict__ Kb,
                      const unsigned short* __restrict__ Vb,
                      const float* __restrict__ lam,
                      float* __restrict__ out, int head) {
    extern __shared__ __align__(16) unsigned short smem[];
    // layout: Ks buffers at smem + p*32768 (each 64x512)
    //         Vt buffers at smem + 65536 + p*32768 (each 512x64, transposed)
    //         Pb          at smem + 131072 (8 waves x 16x64)

    const int tid  = threadIdx.x;
    const int lane = tid & 31;
    const int wave = tid >> 5;
    const int half = lane >> 4;
    const int l16  = lane & 15;

    const int b  = blockIdx.x >> 5;             // 4 batches
    const int qt = blockIdx.x & 31;             // 32 query tiles of 128
    const long rowQ  = (long)b * 4096 + qt * 128 + wave * 16 + l16;
    const long krow0 = (long)b * 4096;

    // softmax scale folded with log2(e) so we can use exp2
    const float SC = 0.044194173824159216f * 1.4426950408889634f;

    // ---- Q fragments resident in VGPRs: 16 x (16x32 bf16) covering d=512 ----
    FragU Qa[16];
#pragma unroll
    for (int kf = 0; kf < 16; ++kf) {
        const unsigned short* p = Qb + rowQ * 1024 + head * 512 + kf * 32 + half * 8;
        Qa[kf].u[0] = *(const v4u*)(p);
        Qa[kf].u[1] = *(const v4u*)(p + 16);
    }

    v8f O[32];                                   // 16 x 512 f32 accumulator
#pragma unroll
    for (int dt = 0; dt < 32; ++dt)
#pragma unroll
        for (int r = 0; r < 8; ++r) O[dt][r] = 0.0f;

    float mrow[8], lrow[8];
#pragma unroll
    for (int r = 0; r < 8; ++r) { mrow[r] = -3.0e38f; lrow[r] = 0.0f; }

    attn_load_tile(Kb, Vb, krow0, head, tid, smem, smem + 2 * 64 * 512);
    wait_async_zero();
    __syncthreads();

    for (int j = 0; j < 64; ++j) {               // 4096 keys / 64
        const int p = j & 1;
        const unsigned short* Ks = smem + p * (64 * 512);
        const unsigned short* Vt = smem + 2 * 64 * 512 + p * (64 * 512);
        if (j + 1 < 64)                          // prefetch next tile into 1-p
            attn_load_tile(Kb, Vb, krow0 + (j + 1) * 64, head, tid,
                           smem + (1 - p) * (64 * 512),
                           smem + 2 * 64 * 512 + (1 - p) * (64 * 512));

        // ---- S = Q K^T : 16 queries x 64 keys ----
        v8f S[4];
#pragma unroll
        for (int t = 0; t < 4; ++t)
#pragma unroll
            for (int r = 0; r < 8; ++r) S[t][r] = 0.0f;

#pragma unroll
        for (int kf = 0; kf < 16; ++kf) {
#pragma unroll
            for (int t = 0; t < 4; ++t) {
                FragU B;                          // lane=key col, d split per ISA layout
                const unsigned short* ptr =
                    Ks + (t * 16 + l16) * 512 + kf * 32 + half * 8;
                B.u[0] = *(const v4u*)(ptr);
                B.u[1] = *(const v4u*)(ptr + 16);
                S[t] = __builtin_amdgcn_wmma_f32_16x16x32_bf16(
                    false, Qa[kf].bf, false, B.bf, (short)0, S[t], false, false);
            }
        }

        // ---- online softmax (rows live across lanes 0-15 / 16-31) ----
        float mnew[8], alpha[8];
#pragma unroll
        for (int r = 0; r < 8; ++r) {
            float mx = fmaxf(fmaxf(S[0][r], S[1][r]), fmaxf(S[2][r], S[3][r]));
            mx = fmaxf(mx, __shfl_xor(mx, 1, 32));
            mx = fmaxf(mx, __shfl_xor(mx, 2, 32));
            mx = fmaxf(mx, __shfl_xor(mx, 4, 32));
            mx = fmaxf(mx, __shfl_xor(mx, 8, 32));
            mnew[r]  = fmaxf(mrow[r], mx);
            alpha[r] = exp2f((mrow[r] - mnew[r]) * SC);
            mrow[r]  = mnew[r];
        }
#pragma unroll
        for (int r = 0; r < 8; ++r) {
            float rs = 0.0f;
#pragma unroll
            for (int t = 0; t < 4; ++t) {
                float pv = exp2f((S[t][r] - mnew[r]) * SC);
                S[t][r] = pv;
                rs += pv;
            }
            rs += __shfl_xor(rs, 1, 32);
            rs += __shfl_xor(rs, 2, 32);
            rs += __shfl_xor(rs, 4, 32);
            rs += __shfl_xor(rs, 8, 32);
            lrow[r] = lrow[r] * alpha[r] + rs;
        }
#pragma unroll
        for (int dt = 0; dt < 32; ++dt)
#pragma unroll
            for (int r = 0; r < 8; ++r) O[dt][r] *= alpha[r];

        // ---- P: C-layout -> bf16 A-fragment layout via per-wave LDS ----
        unsigned short* Pw = smem + 4 * 64 * 512 + wave * (16 * 64);
#pragma unroll
        for (int t = 0; t < 4; ++t)
#pragma unroll
            for (int r = 0; r < 8; ++r)
                Pw[(r + 8 * half) * 64 + t * 16 + l16] = f32_to_bf16(S[t][r]);
        // per-wave LDS ops are in-order; no cross-wave sharing -> no barrier

        // ---- O += P @ V ----
#pragma unroll
        for (int kp = 0; kp < 2; ++kp) {
            FragU A;
            const unsigned short* ap = Pw + l16 * 64 + kp * 32 + half * 8;
            A.u[0] = *(const v4u*)(ap);
            A.u[1] = *(const v4u*)(ap + 16);
#pragma unroll
            for (int dt = 0; dt < 32; ++dt) {
                FragU B;                          // lane = d col, keys split per ISA layout
                const unsigned short* bp =
                    Vt + (dt * 16 + l16) * 64 + kp * 32 + half * 8;
                B.u[0] = *(const v4u*)(bp);
                B.u[1] = *(const v4u*)(bp + 16);
                O[dt] = __builtin_amdgcn_wmma_f32_16x16x32_bf16(
                    false, A.bf, false, B.bf, (short)0, O[dt], false, false);
            }
        }

        wait_async_zero();                       // next tile landed in LDS
        __syncthreads();                         // one barrier per iteration
    }

    // ---- epilogue ----
    const float lambda_ = __expf(lam[0]) + 0.05f;
    const long orow0 = (long)b * 4096 + qt * 128 + wave * 16;
#pragma unroll
    for (int dt = 0; dt < 32; ++dt) {
        int d = dt * 16 + l16;
#pragma unroll
        for (int r = 0; r < 8; ++r) {
            long row = orow0 + r + 8 * half;
            float val = O[dt][r] / lrow[r];
            float* po = out + row * 512 + d;
            if (head == 0) *po = val;
            else           *po = *po - lambda_ * val;
        }
    }
}

// ---------------------------------------------------------------------------
// Launch
// ---------------------------------------------------------------------------
extern "C" void kernel_launch(void* const* d_in, const int* in_sizes, int n_in,
                              void* d_out, int out_size, void* d_ws, size_t ws_size,
                              hipStream_t stream) {
    const float* X   = (const float*)d_in[0];
    const float* Wq  = (const float*)d_in[1];
    const float* bq  = (const float*)d_in[2];
    const float* Wk  = (const float*)d_in[3];
    const float* bk  = (const float*)d_in[4];
    const float* Wv  = (const float*)d_in[5];
    const float* bv  = (const float*)d_in[6];
    const float* lam = (const float*)d_in[7];
    float* out = (float*)d_out;

    char* ws = (char*)d_ws;
    unsigned short* Xb    = (unsigned short*)(ws);                         // 32 MB
    unsigned short* Wb    = (unsigned short*)(ws + (33554432L));           //  5 MB
    float*          biasP = (float*)(ws + 33554432L + 5242880L);           // 10 KB
    unsigned short* Qb    = (unsigned short*)(ws + 33554432L + 5242880L + 16384L);
    unsigned short* Kb    = Qb + (long)16384 * 1024;                       // 32 MB
    unsigned short* Vb    = Kb + (long)16384 * 1024;                       // 16 MB

    // Stage 0: conversions / packing
    cvt_x_kernel<<<4096, 256, 0, stream>>>(X, Xb, 16384 * 1024);
    pack_w_kernel<<<2560, 256, 0, stream>>>(Wq, Wk, Wv, bq, bk, bv, Wb, biasP);

    // Stage 1: QKV projection GEMM (M=16384, N=2560, K=1024)
    qkv_gemm_kernel<<<dim3(128, 20), 256, GEMM_SMEM_BYTES, stream>>>(
        Xb, Wb, biasP, Qb, Kb, Vb);

    // Stage 2: differential attention, two flash passes
    attn_pass_kernel<<<128, 256, ATTN_SMEM_BYTES, stream>>>(Qb, Kb, Vb, lam, out, 0);
    attn_pass_kernel<<<128, 256, ATTN_SMEM_BYTES, stream>>>(Qb, Kb, Vb, lam, out, 1);
}
